// MultiheadAttention_63118839382212
// MI455X (gfx1250) — matrix operand order repaired and checked by reading, not compiled, not run on hardware
//
#include <hip/hip_runtime.h>
#include <hip/hip_bf16.h>
#include <math.h>

#define B_    4
#define CIN   256
#define HEADS 8
#define HD    64
#define NN    2304        // 48*48
#define HB    32          // HEADS*B
#define COUT  512         // HEADS*HD
#define SCALE 0.125f      // 1/sqrt(64)

typedef __attribute__((ext_vector_type(16))) _Float16 v16h;
typedef __attribute__((ext_vector_type(8)))  float    v8f;

union Frag {
  v16h     v;
  uint32_t u[8];
  _Float16 h[16];
};

// A-matrix (16x32 f16) K base for vgpr i, lane half (contiguous k = 16*half + 2*i, matching B)
__device__ __forceinline__ int kbaseA(int i, int half) {
  return 16 * half + 2 * i;
}

__device__ __forceinline__ v8f wmma16(v16h a, v16h b, v8f c) {
  return __builtin_amdgcn_wmma_f32_16x16x32_f16(false, a, false, b, (short)0, c,
                                                false, false);
}

// ---------------- Stage 1: conversions ----------------

// x [B][C][N] f32 -> x16 [B][N][C] f16 (transposed for A-fragments)
__global__ __launch_bounds__(256) void k_cvt_x(const float* __restrict__ x,
                                               _Float16* __restrict__ x16) {
  int idx = blockIdx.x * 256 + threadIdx.x;          // B*N*C = 2359296 exact
  int c = idx % CIN;
  int t = idx / CIN;
  int n = t % NN;
  int b = t / NN;
  x16[idx] = (_Float16)x[((size_t)b * CIN + c) * NN + n];
}

// Wq|Wk|Wv ([H][HD][C] each) and Wo ([256][512]) -> f16, same layouts
__global__ __launch_bounds__(256) void k_cvt_w(const float* __restrict__ wq,
                                               const float* __restrict__ wk,
                                               const float* __restrict__ wv,
                                               const float* __restrict__ wo,
                                               _Float16* __restrict__ wqkv16,
                                               _Float16* __restrict__ wo16) {
  int idx = blockIdx.x * 256 + threadIdx.x;          // 4*131072 = 524288 exact
  int sel = idx >> 17;                               // 131072 = 1<<17
  int r = idx & 131071;
  if (sel == 0)      wqkv16[r]            = (_Float16)wq[r];
  else if (sel == 1) wqkv16[131072 + r]   = (_Float16)wk[r];
  else if (sel == 2) wqkv16[262144 + r]   = (_Float16)wv[r];
  else               wo16[r]              = (_Float16)wo[r];
}

// ---------------- Stage 2: QKV projections via WMMA ----------------
// q/k: [hb][n][d] f16 ; v: transposed [hb][d][n] f16
__global__ __launch_bounds__(128) void k_proj(const _Float16* __restrict__ x16,
                                              const _Float16* __restrict__ wqkv,
                                              _Float16* __restrict__ q16,
                                              _Float16* __restrict__ k16,
                                              _Float16* __restrict__ v16t) {
  int gw = (blockIdx.x * blockDim.x + threadIdx.x) >> 5;
  int lane = threadIdx.x & 31;
  int m = lane & 15, half = lane >> 4;

  int dtile = gw & 3;  gw >>= 2;            // 4 tiles of d
  int ntile = gw % 144; gw /= 144;          // 144 tiles of n
  int hb = gw & 31;
  int which = gw >> 5;                      // 0=Q 1=K 2=V
  int h = hb >> 2, b = hb & 3;

  const _Float16* ap = x16 + ((size_t)b * NN + ntile * 16 + m) * CIN;
  const _Float16* wp = wqkv + (size_t)which * 131072 +
                       ((size_t)h * HD + dtile * 16 + m) * CIN;

  v8f acc = {};
#pragma unroll
  for (int kk = 0; kk < 8; ++kk) {
    int c0 = kk * 32;
    Frag a, w;
#pragma unroll
    for (int i = 0; i < 8; ++i) {
      a.u[i] = *(const uint32_t*)(ap + c0 + kbaseA(i, half));
      w.u[i] = *(const uint32_t*)(wp + c0 + 16 * half + 2 * i);
    }
    acc = wmma16(a.v, w.v, acc);
  }

  int d = dtile * 16 + m;                   // lane = column d
  if (which == 2) {
#pragma unroll
    for (int r = 0; r < 8; ++r) {
      int n = ntile * 16 + r + 8 * half;
      v16t[((size_t)hb * HD + d) * NN + n] = (_Float16)acc[r];
    }
  } else {
    _Float16* dst = (which == 0) ? q16 : k16;
#pragma unroll
    for (int r = 0; r < 8; ++r) {
      int n = ntile * 16 + r + 8 * half;
      dst[((size_t)hb * NN + n) * HD + d] = (_Float16)acc[r];
    }
  }
}

// ---------------- Stage 3: flash attention ----------------
__global__ __launch_bounds__(128) void k_flash(const _Float16* __restrict__ q16,
                                               const _Float16* __restrict__ k16,
                                               const _Float16* __restrict__ v16t,
                                               _Float16* __restrict__ att16) {
  __shared__ _Float16 lds[4][16 * 32];      // per-wave P tile
  int wv = threadIdx.x >> 5;
  int lane = threadIdx.x & 31;
  int m = lane & 15, half = lane >> 4;
  int gw = blockIdx.x * 4 + wv;
  int ntile = gw % 144;
  int hb = gw / 144;

  // Q A-fragments (d 0..31 and 32..63)
  Frag aq0, aq1;
  const _Float16* qp = q16 + ((size_t)hb * NN + ntile * 16 + m) * HD;
#pragma unroll
  for (int i = 0; i < 8; ++i) {
    int k0 = kbaseA(i, half);
    aq0.u[i] = *(const uint32_t*)(qp + k0);
    aq1.u[i] = *(const uint32_t*)(qp + 32 + k0);
  }

  v8f acc[4] = {{}, {}, {}, {}};
  float mi[8], li[8];
#pragma unroll
  for (int r = 0; r < 8; ++r) { mi[r] = -INFINITY; li[r] = 0.f; }

  const _Float16* kb = k16 + (size_t)hb * NN * HD;
  const _Float16* vb = v16t + (size_t)hb * HD * NN;
  _Float16* pl = lds[wv];

  for (int mj = 0; mj < NN; mj += 32) {
    // K B-fragments: cols mj..mj+15 and mj+16..mj+31, d-halves 0..31 / 32..63
    Frag b00, b01, b10, b11;
    const _Float16* kp0 = kb + ((size_t)mj + m) * HD;
    const _Float16* kp1 = kb + ((size_t)mj + 16 + m) * HD;
#pragma unroll
    for (int i = 0; i < 8; ++i) {
      int d0 = 16 * half + 2 * i;
      b00.u[i] = *(const uint32_t*)(kp0 + d0);
      b01.u[i] = *(const uint32_t*)(kp0 + 32 + d0);
      b10.u[i] = *(const uint32_t*)(kp1 + d0);
      b11.u[i] = *(const uint32_t*)(kp1 + 32 + d0);
    }
    v8f s0 = {}, s1 = {};
    s0 = wmma16(aq0.v, b00.v, s0);
    s0 = wmma16(aq1.v, b01.v, s0);
    s1 = wmma16(aq0.v, b10.v, s1);
    s1 = wmma16(aq1.v, b11.v, s1);

    // online softmax: rows live in (vgpr r, half); cols across 16 lanes
    float pr0[8], pr1[8];
#pragma unroll
    for (int r = 0; r < 8; ++r) {
      float a0 = s0[r] * SCALE, a1 = s1[r] * SCALE;
      float mx = fmaxf(a0, a1);
#pragma unroll
      for (int msk = 1; msk < 16; msk <<= 1)
        mx = fmaxf(mx, __shfl_xor(mx, msk, 32));
      float mnew = fmaxf(mi[r], mx);
      float p0 = __expf(a0 - mnew);
      float p1 = __expf(a1 - mnew);
      float so = __expf(mi[r] - mnew);
      float sum = p0 + p1;
#pragma unroll
      for (int msk = 1; msk < 16; msk <<= 1)
        sum += __shfl_xor(sum, msk, 32);
      li[r] = li[r] * so + sum;
      mi[r] = mnew;
      acc[0][r] *= so; acc[1][r] *= so; acc[2][r] *= so; acc[3][r] *= so;
      pr0[r] = p0; pr1[r] = p1;
    }

    // P: D-layout -> A-layout transpose through per-wave LDS tile [row][col]
#pragma unroll
    for (int r = 0; r < 8; ++r) {
      int row = r + 8 * half;
      pl[row * 32 + m]      = (_Float16)pr0[r];
      pl[row * 32 + 16 + m] = (_Float16)pr1[r];
    }
    Frag ap;
#pragma unroll
    for (int i = 0; i < 8; ++i)
      ap.u[i] = *(const uint32_t*)(pl + m * 32 + kbaseA(i, half));

    // PV: V B-fragments from transposed V (contiguous in m)
#pragma unroll
    for (int t = 0; t < 4; ++t) {
      Frag bv;
      const _Float16* vp = vb + ((size_t)(t * 16 + m)) * NN + mj;
#pragma unroll
      for (int i = 0; i < 8; ++i)
        bv.u[i] = *(const uint32_t*)(vp + 16 * half + 2 * i);
      acc[t] = wmma16(ap.v, bv.v, acc[t]);
    }
  }

  // normalize + store att16 [hb][n][d] f16
#pragma unroll
  for (int r = 0; r < 8; ++r) {
    float inv = 1.f / li[r];
    int n = ntile * 16 + r + 8 * half;
#pragma unroll
    for (int t = 0; t < 4; ++t)
      att16[((size_t)hb * NN + n) * HD + t * 16 + m] =
          (_Float16)(acc[t][r] * inv);
  }
}

// ---------------- Stage 4: torch transpose(1,2)+reshape replication ----------------
// attr [b2][n2][c] (c fastest, for B-fragments) gathered from att16 [h][b][n][d]
__global__ __launch_bounds__(256) void k_rearrange(const _Float16* __restrict__ att16,
                                                   _Float16* __restrict__ attr) {
  int idx = blockIdx.x * 256 + threadIdx.x;   // 4*2304*512 = 4718592 exact
  int c = idx & 511;
  int t = idx >> 9;
  int n2 = t % NN;
  int b2 = t / NN;
  int F = (b2 * COUT + c) * NN + n2;          // flat idx of [B,512,N] view
  int d = F & 63;
  int bb = (F >> 6) & 3;
  int n = (F >> 8) % NN;
  int h = F / (NN * B_ * HD);                 // /589824
  attr[idx] = att16[(((size_t)(h * 4 + bb) * NN + n) * HD + d)];
}

// ---------------- Stage 5: output 1x1 conv via WMMA ----------------
__global__ __launch_bounds__(128) void k_out(const _Float16* __restrict__ wo16,
                                             const _Float16* __restrict__ attr,
                                             float* __restrict__ out) {
  int gw = (blockIdx.x * blockDim.x + threadIdx.x) >> 5;
  int lane = threadIdx.x & 31;
  int m = lane & 15, half = lane >> 4;
  int nt = gw % 144; gw /= 144;
  int ot = gw & 15;
  int b2 = gw >> 4;

  const _Float16* ap = wo16 + ((size_t)(ot * 16 + m)) * COUT;
  const _Float16* bp = attr + ((size_t)b2 * NN + nt * 16 + m) * COUT;

  v8f acc = {};
#pragma unroll
  for (int kk = 0; kk < 16; ++kk) {
    int c0 = kk * 32;
    Frag a, bfr;
#pragma unroll
    for (int i = 0; i < 8; ++i) {
      a.u[i]   = *(const uint32_t*)(ap + c0 + kbaseA(i, half));
      bfr.u[i] = *(const uint32_t*)(bp + c0 + 16 * half + 2 * i);
    }
    acc = wmma16(a.v, bfr.v, acc);
  }
#pragma unroll
  for (int r = 0; r < 8; ++r) {
    int o = ot * 16 + r + 8 * half;
    out[((size_t)(b2 * CIN + o)) * NN + nt * 16 + m] = acc[r];
  }
}

// ---------------- launch ----------------
extern "C" void kernel_launch(void* const* d_in, const int* in_sizes, int n_in,
                              void* d_out, int out_size, void* d_ws, size_t ws_size,
                              hipStream_t stream) {
  (void)in_sizes; (void)n_in; (void)out_size; (void)ws_size;
  const float* x  = (const float*)d_in[0];
  const float* wq = (const float*)d_in[1];
  const float* wk = (const float*)d_in[2];
  const float* wv = (const float*)d_in[3];
  const float* wo = (const float*)d_in[4];
  float* out = (float*)d_out;

  char* p = (char*)d_ws;
  _Float16* x16   = (_Float16*)p; p += (size_t)B_ * NN * CIN * 2;      // 4.72 MB
  _Float16* wqkv  = (_Float16*)p; p += (size_t)3 * 131072 * 2;         // 0.79 MB
  _Float16* wo16  = (_Float16*)p; p += (size_t)131072 * 2;             // 0.26 MB
  _Float16* q16   = (_Float16*)p; p += (size_t)HB * NN * HD * 2;       // 9.44 MB
  _Float16* k16   = (_Float16*)p; p += (size_t)HB * NN * HD * 2;
  _Float16* v16t  = (_Float16*)p; p += (size_t)HB * NN * HD * 2;
  _Float16* att16 = (_Float16*)p; p += (size_t)HB * NN * HD * 2;       // ~43.5 MB total
  _Float16* attr  = q16;  // alias: Q dead after flash, same size as attr

  k_cvt_x<<<(B_ * NN * CIN) / 256, 256, 0, stream>>>(x, x16);
  k_cvt_w<<<(4 * 131072) / 256, 256, 0, stream>>>(wq, wk, wv, wo, wqkv, wo16);
  k_proj<<<13824, 128, 0, stream>>>(x16, wqkv, q16, k16, v16t);   // 55296 waves
  k_flash<<<1152, 128, 0, stream>>>(q16, k16, v16t, att16);       // 4608 waves
  k_rearrange<<<(B_ * NN * COUT) / 256, 256, 0, stream>>>(att16, attr);
  k_out<<<2304, 128, 0, stream>>>(wo16, attr, out);               // 9216 waves
}
